// HierarchicalAggregation_88295937671207
// MI455X (gfx1250) — matrix operand order repaired and checked by reading, not compiled, not run on hardware
//
#include <hip/hip_runtime.h>

typedef __attribute__((ext_vector_type(16))) __bf16 v16bf;
typedef __attribute__((ext_vector_type(8)))  float  v8f;

#define BN_EPS 1e-5f
#define NB 32
#define NG 15
#define NK 2048
#define NPTS_F 983040.0f
#define Y1P 72   // LDS pitch for 64-wide bf16 rows: 144B, 16B-aligned, conflict-free

__constant__ int SIDX[5][3] = {{0,1,8},{2,4,6},{3,5,7},{9,11,13},{10,12,14}};

// ---- workspace layout (float offsets) ----
enum : int {
  WS_ACC    = 0,       // 9 used (sum_r[3], M[6])
  WS_S1     = 16,      // 64
  WS_S2     = 80,      // 64*64
  WS_STAT2A = 4176,    // 256  (sum[128], sumsq[128])
  WS_STAT2B = 4432,    // 256
  WS_STAT3A = 4688,    // 512
  WS_STAT3B = 5200,    // 512
  WS_ZEND   = 5712,    // end of zero-initialized region
  WS_CENT   = 5712,    // 32*15*3   [b][g][i]
  WS_A1C1   = 7152,    // 64*4      folded layer-1 (A1[0..2], c1)
  WS_W1B    = 7408,    // 64*64     folded layer-2 weights
  WS_C2F    = 11504,   // 64        folded layer-2 bias
  WS_LF1    = 11568,   // 32*64*15  [b][c][g]
  WS_C2S    = 42288,   // 32*5*3    [b][s][i]
  WS_X2     = 42768,   // 67*480
  WS_X3     = 74928,   // 131*160
  WS_P      = 95888,   // 61440
  WS_Q      = 157328,  // 61440
  WS_R      = 218768   // 61440
};

#define OUT_LF 8192     // gf is [0,8192), local_features at 8192

// ---------------------------------------------------------------------------
__device__ __forceinline__ float block_reduce_sum(float v, float* red) {
  int t = threadIdx.x;
  red[t] = v; __syncthreads();
  for (int o = 128; o > 0; o >>= 1) { if (t < o) red[t] += red[t + o]; __syncthreads(); }
  float r = red[0]; __syncthreads();
  return r;
}

// Pass 1: centroids per (b,g); global 3-vector sum + 3x3 second moment of r.
__global__ void s1_pass1(const float* __restrict__ lr, float* __restrict__ ws,
                         float* __restrict__ out) {
  __shared__ float red[256];
  int bg = blockIdx.x, b = bg / NG, g = bg % NG, t = threadIdx.x;
  const float* p = lr + (size_t)bg * NK * 3;
  float sx = 0.f, sy = 0.f, sz = 0.f;
  for (int k = t; k < NK; k += 256) {
    sx += p[k*3+0]; sy += p[k*3+1]; sz += p[k*3+2];
  }
  float cx = block_reduce_sum(sx, red) * (1.0f/NK);
  float cy = block_reduce_sum(sy, red) * (1.0f/NK);
  float cz = block_reduce_sum(sz, red) * (1.0f/NK);
  float a[9]; for (int i = 0; i < 9; i++) a[i] = 0.f;
  for (int k = t; k < NK; k += 256) {
    float r0 = p[k*3+0]-cx, r1 = p[k*3+1]-cy, r2 = p[k*3+2]-cz;
    a[0]+=r0; a[1]+=r1; a[2]+=r2;
    a[3]+=r0*r0; a[4]+=r0*r1; a[5]+=r0*r2;
    a[6]+=r1*r1; a[7]+=r1*r2; a[8]+=r2*r2;
  }
  for (int i = 0; i < 9; i++) {
    float r = block_reduce_sum(a[i], red);
    if (t == 0) atomicAdd(&ws[WS_ACC + i], r);
  }
  if (t == 0) {
    ws[WS_CENT + bg*3+0] = cx; ws[WS_CENT + bg*3+1] = cy; ws[WS_CENT + bg*3+2] = cz;
    out[OUT_LF + (b*67 + 0)*NG + g] = cx;
    out[OUT_LF + (b*67 + 1)*NG + g] = cy;
    out[OUT_LF + (b*67 + 2)*NG + g] = cz;
  }
}

// Fold BN1 analytically into layer-1 weights: y1 = relu(A1*r + c1)
__global__ void fold1(const float* __restrict__ W0, const float* __restrict__ gm,
                      const float* __restrict__ bt, float* __restrict__ ws) {
  int c = threadIdx.x;
  float n = NPTS_F;
  float m0 = ws[WS_ACC+0]/n, m1 = ws[WS_ACC+1]/n, m2 = ws[WS_ACC+2]/n;
  float C00 = ws[WS_ACC+3]/n - m0*m0, C01 = ws[WS_ACC+4]/n - m0*m1;
  float C02 = ws[WS_ACC+5]/n - m0*m2, C11 = ws[WS_ACC+6]/n - m1*m1;
  float C12 = ws[WS_ACC+7]/n - m1*m2, C22 = ws[WS_ACC+8]/n - m2*m2;
  float w0 = W0[c*3+0], w1 = W0[c*3+1], w2 = W0[c*3+2];
  float mu  = w0*m0 + w1*m1 + w2*m2;
  float var = w0*w0*C00 + w1*w1*C11 + w2*w2*C22
            + 2.f*(w0*w1*C01 + w0*w2*C02 + w1*w2*C12);
  float inv = gm[c] * rsqrtf(var + BN_EPS);
  ws[WS_A1C1 + c*4+0] = inv*w0;
  ws[WS_A1C1 + c*4+1] = inv*w1;
  ws[WS_A1C1 + c*4+2] = inv*w2;
  ws[WS_A1C1 + c*4+3] = bt[c] - inv*mu;
}

// Build one 64-point chunk of y1 (bf16) in LDS at y1s[c*cs + pt*ps].
// Optionally accumulate fp32 per-thread S1 partials (16 channels/thread).
template<bool WITH_S1>
__device__ __forceinline__ void compute_y1_chunk(
    const float* __restrict__ p, int k0p, const float* __restrict__ a1s,
    float cx, float cy, float cz, __bf16* __restrict__ y1s, int cs, int ps,
    float* __restrict__ ptb, float* s1reg) {
  int t = threadIdx.x;
  if (t < 192) ptb[t] = p[k0p*3 + t];
  __syncthreads();
  int pt = t & 63, cg = t >> 6;
  float r0 = ptb[pt*3+0]-cx, r1 = ptb[pt*3+1]-cy, r2 = ptb[pt*3+2]-cz;
#pragma unroll
  for (int cc = 0; cc < 16; cc++) {
    int c = cg*16 + cc;
    float v = fmaxf(a1s[c*4]*r0 + a1s[c*4+1]*r1 + a1s[c*4+2]*r2 + a1s[c*4+3], 0.f);
    if (WITH_S1) s1reg[cc] += v;
    y1s[c*cs + pt*ps] = (__bf16)v;
  }
  __syncthreads();
}

// Fragment loaders for V_WMMA_F32_16X16X32_BF16 (ISA 16-bit operand layouts).
// A 16x32: lane row=l&15; e<8 -> K = k0+8*hl+e ; e>=8 -> K = k0+16+8*hl+(e-8)
__device__ __forceinline__ v16bf load_afrag(const __bf16* __restrict__ rowbase,
                                            int k0, int hl) {
  v16bf a;
#pragma unroll
  for (int e = 0; e < 8; e++) a[e]     = rowbase[k0 + 8*hl + e];
#pragma unroll
  for (int e = 0; e < 8; e++) a[8 + e] = rowbase[k0 + 16 + 8*hl + e];
  return a;
}
// B 32x16: lane col=l&15; e<8 -> K = k0+8*hl+e ; e>=8 -> K = k0+16+8*hl+(e-8)
__device__ __forceinline__ v16bf load_bfrag(const __bf16* __restrict__ colbase,
                                            int k0, int hl) {
  v16bf b;
#pragma unroll
  for (int e = 0; e < 16; e++) b[e] = colbase[k0 + 8*hl + (e & 7) + ((e >> 3) << 4)];
  return b;
}

// Pass 2: S1 = sum(y1), S2 = Y1*Y1^T via bf16 WMMA (K = points, fp32 accum).
__global__ void s1_pass2(const float* __restrict__ lr, float* __restrict__ ws) {
  __shared__ __align__(16) __bf16 y1s[64 * Y1P];   // [ch][pt]
  __shared__ float a1s[256];
  __shared__ float ptb[192];
  __shared__ float s1b[64];
  int t = threadIdx.x, bg = blockIdx.x;
  const float* p = lr + (size_t)bg * NK * 3;
  a1s[t] = ws[WS_A1C1 + t];
  if (t < 64) s1b[t] = 0.f;
  float cx = ws[WS_CENT + bg*3+0], cy = ws[WS_CENT + bg*3+1], cz = ws[WS_CENT + bg*3+2];
  __syncthreads();
  int lane = t & 31, wave = t >> 5, hl = lane >> 4;
  int m0  = (wave >> 1) * 16;
  int n0A = ((2*wave) & 3) * 16, n0B = n0A + 16;
  int row = lane & 15;
  v8f accA = {0.f,0.f,0.f,0.f,0.f,0.f,0.f,0.f};
  v8f accB = {0.f,0.f,0.f,0.f,0.f,0.f,0.f,0.f};
  float s1reg[16];
#pragma unroll
  for (int i = 0; i < 16; i++) s1reg[i] = 0.f;
  for (int ch = 0; ch < 32; ch++) {
    int k0p = ch * 64;
    if (ch + 1 < 32) __builtin_prefetch(p + (k0p + 64)*3, 0, 1);
    compute_y1_chunk<true>(p, k0p, a1s, cx, cy, cz, y1s, Y1P, 1, ptb, s1reg);
#pragma unroll
    for (int k0 = 0; k0 < 64; k0 += 32) {   // K = point index within chunk
      v16bf a  = load_afrag(y1s + (m0  + row)*Y1P, k0, hl);
      v16bf b0 = load_bfrag(y1s + (n0A + row)*Y1P, k0, hl);
      v16bf b1 = load_bfrag(y1s + (n0B + row)*Y1P, k0, hl);
      accA = __builtin_amdgcn_wmma_f32_16x16x32_bf16(false, a, false, b0, (short)0, accA, false, false);
      accB = __builtin_amdgcn_wmma_f32_16x16x32_bf16(false, a, false, b1, (short)0, accB, false, false);
    }
    __syncthreads();
  }
  { // reduce S1 across points
    int cg = t >> 6;
#pragma unroll
    for (int cc = 0; cc < 16; cc++) atomicAdd(&s1b[cg*16+cc], s1reg[cc]);
  }
  __syncthreads();
  if (t < 64) atomicAdd(&ws[WS_S1 + t], s1b[t]);
#pragma unroll
  for (int v = 0; v < 8; v++) {
    int r = m0 + v + 8*hl;
    atomicAdd(&ws[WS_S2 + r*64 + n0A + row], accA[v]);
    atomicAdd(&ws[WS_S2 + r*64 + n0B + row], accB[v]);
  }
}

// Fold BN2 into layer-2: y2 = relu(W1b*y1 + c2f)
__global__ void fold2(const float* __restrict__ W1, const float* __restrict__ gm,
                      const float* __restrict__ bt, float* __restrict__ ws) {
  __shared__ float my[64];
  int c = threadIdx.x;
  float n = NPTS_F;
  my[c] = ws[WS_S1 + c] / n;
  __syncthreads();
  float mu = 0.f;
  for (int j = 0; j < 64; j++) mu += W1[c*64+j] * my[j];
  float e2 = 0.f;
  for (int j = 0; j < 64; j++) {
    float inner = 0.f;
    for (int k = 0; k < 64; k++) inner += W1[c*64+k] * ws[WS_S2 + j*64 + k];
    e2 += W1[c*64+j] * inner;
  }
  e2 /= n;
  float var = e2 - mu*mu;
  float inv = gm[c] * rsqrtf(var + BN_EPS);
  for (int j = 0; j < 64; j++) ws[WS_W1B + c*64 + j] = inv * W1[c*64+j];
  ws[WS_C2F + c] = bt[c] - inv*mu;
}

// Pass 3: y2 = W1b * y1 via bf16 WMMA (K = channels), +bias, relu, max over points.
__global__ void s1_pass3(const float* __restrict__ lr, float* __restrict__ ws,
                         float* __restrict__ out) {
  __shared__ __align__(16) __bf16 y1t[64 * Y1P];   // [pt][ch] (transposed)
  __shared__ __align__(16) __bf16 w1bs[64 * Y1P];  // [out][in]
  __shared__ float a1s[256];
  __shared__ float c2fs[64];
  __shared__ float ptb[192];
  __shared__ unsigned cmax[64];
  int t = threadIdx.x, bg = blockIdx.x, b = bg / NG, g = bg % NG;
  const float* p = lr + (size_t)bg * NK * 3;
  a1s[t] = ws[WS_A1C1 + t];
  for (int i = t; i < 4096; i += 256)
    w1bs[(i >> 6)*Y1P + (i & 63)] = (__bf16)ws[WS_W1B + i];
  if (t < 64) { c2fs[t] = ws[WS_C2F + t]; cmax[t] = 0u; }
  float cx = ws[WS_CENT + bg*3+0], cy = ws[WS_CENT + bg*3+1], cz = ws[WS_CENT + bg*3+2];
  __syncthreads();
  int lane = t & 31, wave = t >> 5, hl = lane >> 4;
  int m0  = (wave >> 1) * 16;
  int n0A = ((2*wave) & 3) * 16, n0B = n0A + 16;
  int row = lane & 15;
  float rmA[8], rmB[8];
#pragma unroll
  for (int v = 0; v < 8; v++) { rmA[v] = 0.f; rmB[v] = 0.f; }
  for (int ch = 0; ch < 32; ch++) {
    int k0p = ch * 64;
    if (ch + 1 < 32) __builtin_prefetch(p + (k0p + 64)*3, 0, 1);
    compute_y1_chunk<false>(p, k0p, a1s, cx, cy, cz, y1t, 1, Y1P, ptb, nullptr);
    v8f accA = {0.f,0.f,0.f,0.f,0.f,0.f,0.f,0.f};
    v8f accB = {0.f,0.f,0.f,0.f,0.f,0.f,0.f,0.f};
#pragma unroll
    for (int k0 = 0; k0 < 64; k0 += 32) {   // K = input channel
      v16bf a  = load_afrag(w1bs + (m0 + row)*Y1P, k0, hl);
      v16bf b0 = load_bfrag(y1t + (n0A + row)*Y1P, k0, hl);  // col = point
      v16bf b1 = load_bfrag(y1t + (n0B + row)*Y1P, k0, hl);
      accA = __builtin_amdgcn_wmma_f32_16x16x32_bf16(false, a, false, b0, (short)0, accA, false, false);
      accB = __builtin_amdgcn_wmma_f32_16x16x32_bf16(false, a, false, b1, (short)0, accB, false, false);
    }
#pragma unroll
    for (int v = 0; v < 8; v++) {
      int chn = m0 + v + 8*hl;
      rmA[v] = fmaxf(rmA[v], fmaxf(accA[v] + c2fs[chn], 0.f));
      rmB[v] = fmaxf(rmB[v], fmaxf(accB[v] + c2fs[chn], 0.f));
    }
    __syncthreads();
  }
#pragma unroll
  for (int v = 0; v < 8; v++) {  // relu output >= 0: float order == uint order
    int chn = m0 + v + 8*hl;
    atomicMax(&cmax[chn], __float_as_uint(rmA[v]));
    atomicMax(&cmax[chn], __float_as_uint(rmB[v]));
  }
  __syncthreads();
  if (t < 64) {
    float m = __uint_as_float(cmax[t]);
    ws[WS_LF1 + (b*64 + t)*NG + g] = m;
    out[OUT_LF + (b*67 + 3 + t)*NG + g] = m;
  }
}

// Build stage-2 input X2 (67x480), c2, and X3 coordinate rows (rel3).
__global__ void build2(float* __restrict__ ws) {
  __shared__ float c2l[5][3];
  __shared__ float c3l[3];
  int b = blockIdx.x, t = threadIdx.x;
  if (t < 15) {
    int s = t / 3, i = t % 3;
    float sum = 0.f;
    for (int j = 0; j < 3; j++) sum += ws[WS_CENT + (b*NG + SIDX[s][j])*3 + i];
    float m = sum * (1.f/3.f);
    c2l[s][i] = m;
    ws[WS_C2S + (b*5 + s)*3 + i] = m;
  }
  __syncthreads();
  if (t < 3) c3l[t] = (c2l[0][t]+c2l[1][t]+c2l[2][t]+c2l[3][t]+c2l[4][t]) * 0.2f;
  __syncthreads();
  if (t < 45) {  // rel2 -> X2 rows 0..2
    int s = t / 9, j = (t % 9) / 3, i = t % 3;
    int col = b*15 + s*3 + j;
    float cen = ws[WS_CENT + (b*NG + SIDX[s][j])*3 + i];
    ws[WS_X2 + i*480 + col] = cen - c2l[s][i];
  }
  if (t < 15) {  // rel3 -> X3 rows 0..2
    int s = t / 3, i = t % 3;
    ws[WS_X3 + i*160 + b*5 + s] = c2l[s][i] - c3l[i];
  }
  // lf1_s -> X2 rows 3..66 (thread t = channel)
  for (int s = 0; s < 5; s++)
    for (int j = 0; j < 3; j++)
      ws[WS_X2 + (3 + t)*480 + b*15 + s*3 + j] =
          ws[WS_LF1 + (b*64 + t)*NG + SIDX[s][j]];
}

// Generic tiny GEMM T = W*X with per-row sum/sumsq stats.
__global__ void gemm_stats(const float* __restrict__ W, const float* __restrict__ X,
                           float* __restrict__ T, float* __restrict__ stat,
                           int Cout, int Cin, int N) {
  int id = blockIdx.x * 256 + threadIdx.x;
  if (id >= Cout * N) return;
  int row = id / N, col = id % N;
  float acc = 0.f;
  for (int i = 0; i < Cin; i++) acc += W[row*Cin + i] * X[i*N + col];
  T[id] = acc;
  atomicAdd(&stat[row], acc);
  atomicAdd(&stat[Cout + row], acc * acc);
}

__global__ void bnrelu(const float* __restrict__ T, float* __restrict__ Y,
                       const float* __restrict__ stat, const float* __restrict__ gm,
                       const float* __restrict__ bt, int Cout, int N) {
  int id = blockIdx.x * 256 + threadIdx.x;
  if (id >= Cout * N) return;
  int row = id / N;
  float mu  = stat[row] / N;
  float var = stat[Cout + row] / N - mu*mu;
  Y[id] = fmaxf(gm[row]*(T[id]-mu)*rsqrtf(var + BN_EPS) + bt[row], 0.f);
}

__global__ void max2(const float* __restrict__ Y, float* __restrict__ ws) {
  int id = blockIdx.x * 256 + threadIdx.x;       // (c, b, s) over 128*160
  if (id >= 128*160) return;
  int c = id / 160, bs = id % 160, b = bs / 5, s = bs % 5;
  const float* p = Y + c*480 + b*15 + s*3;
  ws[WS_X3 + (3 + c)*160 + bs] = fmaxf(fmaxf(p[0], p[1]), p[2]);
}

__global__ void max3(const float* __restrict__ Y, float* __restrict__ out) {
  int id = blockIdx.x * 256 + threadIdx.x;       // (b, c) over 32*256
  if (id >= 32*256) return;
  int b = id / 256, c = id % 256;
  const float* p = Y + c*160 + b*5;
  float m = p[0];
  for (int s = 1; s < 5; s++) m = fmaxf(m, p[s]);
  out[b*256 + c] = m;
}

// ---------------------------------------------------------------------------
extern "C" void kernel_launch(void* const* d_in, const int* in_sizes, int n_in,
                              void* d_out, int out_size, void* d_ws, size_t ws_size,
                              hipStream_t stream) {
  const float* lr     = (const float*)d_in[0];
  const float* la0_w0 = (const float*)d_in[1];
  const float* la0_g0 = (const float*)d_in[2];
  const float* la0_b0 = (const float*)d_in[3];
  const float* la0_w1 = (const float*)d_in[4];
  const float* la0_g1 = (const float*)d_in[5];
  const float* la0_b1 = (const float*)d_in[6];
  const float* la1_w0 = (const float*)d_in[7];
  const float* la1_g0 = (const float*)d_in[8];
  const float* la1_b0 = (const float*)d_in[9];
  const float* la1_w1 = (const float*)d_in[10];
  const float* la1_g1 = (const float*)d_in[11];
  const float* la1_b1 = (const float*)d_in[12];
  const float* la2_w0 = (const float*)d_in[13];
  const float* la2_g0 = (const float*)d_in[14];
  const float* la2_b0 = (const float*)d_in[15];
  const float* la2_w1 = (const float*)d_in[16];
  const float* la2_g1 = (const float*)d_in[17];
  const float* la2_b1 = (const float*)d_in[18];
  float* out = (float*)d_out;
  float* ws  = (float*)d_ws;

  hipMemsetAsync(d_ws, 0, (size_t)WS_ZEND * sizeof(float), stream);

  // Stage 1 (the heavy part): analytic-BN folded MLP with bf16 WMMA GEMMs.
  s1_pass1<<<NB*NG, 256, 0, stream>>>(lr, ws, out);
  fold1<<<1, 64, 0, stream>>>(la0_w0, la0_g0, la0_b0, ws);
  s1_pass2<<<NB*NG, 256, 0, stream>>>(lr, ws);
  fold2<<<1, 64, 0, stream>>>(la0_w1, la0_g1, la0_b1, ws);
  s1_pass3<<<NB*NG, 256, 0, stream>>>(lr, ws, out);

  // Stage 2 (tiny): 67->128->128 over 480 columns, max over 3.
  build2<<<NB, 64, 0, stream>>>(ws);
  gemm_stats<<<240, 256, 0, stream>>>(la1_w0, ws+WS_X2, ws+WS_P, ws+WS_STAT2A, 128, 67, 480);
  bnrelu<<<240, 256, 0, stream>>>(ws+WS_P, ws+WS_Q, ws+WS_STAT2A, la1_g0, la1_b0, 128, 480);
  gemm_stats<<<240, 256, 0, stream>>>(la1_w1, ws+WS_Q, ws+WS_P, ws+WS_STAT2B, 128, 128, 480);
  bnrelu<<<240, 256, 0, stream>>>(ws+WS_P, ws+WS_R, ws+WS_STAT2B, la1_g1, la1_b1, 128, 480);
  max2<<<80, 256, 0, stream>>>(ws+WS_R, ws);

  // Stage 3 (tiny): 131->256->256 over 160 columns, max over 5 -> gf.
  gemm_stats<<<160, 256, 0, stream>>>(la2_w0, ws+WS_X3, ws+WS_P, ws+WS_STAT3A, 256, 131, 160);
  bnrelu<<<160, 256, 0, stream>>>(ws+WS_P, ws+WS_Q, ws+WS_STAT3A, la2_g0, la2_b0, 256, 160);
  gemm_stats<<<160, 256, 0, stream>>>(la2_w1, ws+WS_Q, ws+WS_P, ws+WS_STAT3B, 256, 256, 160);
  bnrelu<<<160, 256, 0, stream>>>(ws+WS_P, ws+WS_R, ws+WS_STAT3B, la2_g1, la2_b1, 256, 160);
  max3<<<32, 256, 0, stream>>>(ws+WS_R, out);
}
